// GCN_7035156431047
// MI455X (gfx1250) — compile-verified, matrix operand order checked by reading, not run on hardware
//
#include <hip/hip_runtime.h>
#include <hip/hip_bf16.h>

// ---------------------------------------------------------------------------
// GCN: ChebConv(K=2) x2 + Linear head, MI455X (gfx1250, wave32, WMMA bf16)
// ---------------------------------------------------------------------------

typedef __attribute__((ext_vector_type(16))) __bf16 v16bf;
typedef __attribute__((ext_vector_type(8)))  float  v8f;

#define F_IN   128
#define F_HID  100
#define F_GCN  128
#define F_TR   512
#define LD     128   // padded row stride for all node buffers

__device__ __forceinline__ unsigned short f2bf(float f) {
  union { float f; unsigned u; } x; x.f = f;
  unsigned u = x.u;
  unsigned r = u + 0x7FFFu + ((u >> 16) & 1u);   // round-to-nearest-even
  return (unsigned short)(r >> 16);
}

union FragA { v16bf v; unsigned u[8]; };
union FragC { v8f  v;  float    f[8]; };

// ---------------------------------------------------------------------------
// Tiled GEMM, C[M x Ncols] = A[M x 128] * W[128 x Ncols] via WMMA bf16.
// Block tile: 128(M) x 32(N), 8 waves (one 16-row strip each, 2 N-tiles).
// Weights staged in LDS pre-packed in K-pairs so B fragments are aligned
// 4-byte ds_reads. K is always 128 (narrow dims zero-padded).
// MODE 0 (dual-out): O0 = A0*W0 ; O1 = A0*W1            (layer-1 projections)
// MODE 1 (dual-in) : O0 = act(A0*W0 + A1*W1 + bias)     (layer-2)
// MODE 2 (single)  : O0 = A0*W0 + bias                  (head)
// ---------------------------------------------------------------------------
template <int MODE>
__global__ __launch_bounds__(256) void gemm_wmma(
    const float* __restrict__ A0, const float* __restrict__ A1,
    const float* __restrict__ W0, const float* __restrict__ W1,
    const float* __restrict__ bias,
    float* __restrict__ O0, float* __restrict__ O1,
    int M, int ldW, int Kvalid, int Nvalid, int ldO, int doRelu)
{
  __shared__ __align__(16) unsigned short As0[128 * 32];
  __shared__ __align__(16) unsigned short As1[128 * 32];
  __shared__ __align__(16) unsigned short Bs0[128 * 32]; // (k>>1)*64 + 2*j + (k&1)
  __shared__ __align__(16) unsigned short Bs1[128 * 32];

  const int tid  = threadIdx.x;
  const int wave = tid >> 5;
  const int lane = tid & 31;
  const int half = lane >> 4;
  const int lm   = lane & 15;
  const int m0   = blockIdx.x * 128;
  const int n0   = blockIdx.y * 32;

  // Stage full-K weight tile(s), zero-padded, packed in K-pairs.
  for (int e = tid; e < 128 * 32; e += 256) {
    const int k = e >> 5;            // 0..127
    const int j = e & 31;            // 0..31
    const int n = n0 + j;
    const int dst = ((k >> 1) << 6) + (j << 1) + (k & 1);
    const bool ok = (k < Kvalid) && (n < Nvalid);
    Bs0[dst] = f2bf(ok ? W0[(size_t)k * ldW + n] : 0.f);
    if (MODE != 2)
      Bs1[dst] = f2bf(ok ? W1[(size_t)k * ldW + n] : 0.f);
  }

  v8f acc00 = {}; v8f acc01 = {};   // output0, N-tile 0/1
  v8f acc10 = {}; v8f acc11 = {};   // output1 (MODE 0 only)

  const int rrow = ((wave << 4) + lm) << 5;   // A-fragment row base in LDS

#pragma unroll
  for (int s = 0; s < 4; ++s) {
    const int ks = s * 32;
    __syncthreads();
    // Stage A tile(s) for this K step (coalesced, f32 -> bf16).
    for (int e = tid; e < 128 * 32; e += 256) {
      const int r = e >> 5, kk = e & 31;
      const int m = m0 + r;
      As0[(r << 5) + kk] = f2bf((m < M) ? A0[(size_t)m * LD + ks + kk] : 0.f);
      if (MODE == 1)
        As1[(r << 5) + kk] = f2bf((m < M) ? A1[(size_t)m * LD + ks + kk] : 0.f);
    }
    __syncthreads();

    FragA a0, a1, b00, b01, b10, b11;
#pragma unroll
    for (int v = 0; v < 8; ++v) {
      const int kb = ((v >> 2) << 4) + (half << 3) + ((v & 3) << 1);
      a0.u[v] = *(const unsigned*)&As0[rrow + kb];
      if (MODE == 1) a1.u[v] = *(const unsigned*)&As1[rrow + kb];
      const int boff = (((ks + kb) >> 1) << 6);
      b00.u[v] = *(const unsigned*)&Bs0[boff + (lm << 1)];
      b01.u[v] = *(const unsigned*)&Bs0[boff + ((16 + lm) << 1)];
      if (MODE != 2) {
        b10.u[v] = *(const unsigned*)&Bs1[boff + (lm << 1)];
        b11.u[v] = *(const unsigned*)&Bs1[boff + ((16 + lm) << 1)];
      }
    }
    acc00 = __builtin_amdgcn_wmma_f32_16x16x32_bf16(false, a0.v, false, b00.v, (short)0, acc00, false, false);
    acc01 = __builtin_amdgcn_wmma_f32_16x16x32_bf16(false, a0.v, false, b01.v, (short)0, acc01, false, false);
    if (MODE == 0) {
      acc10 = __builtin_amdgcn_wmma_f32_16x16x32_bf16(false, a0.v, false, b10.v, (short)0, acc10, false, false);
      acc11 = __builtin_amdgcn_wmma_f32_16x16x32_bf16(false, a0.v, false, b11.v, (short)0, acc11, false, false);
    } else if (MODE == 1) {
      acc00 = __builtin_amdgcn_wmma_f32_16x16x32_bf16(false, a1.v, false, b10.v, (short)0, acc00, false, false);
      acc01 = __builtin_amdgcn_wmma_f32_16x16x32_bf16(false, a1.v, false, b11.v, (short)0, acc01, false, false);
    }
  }

  // Store: D VGPR v -> M = v + 8*half, lanes hold N.
  FragC c00, c01, c10, c11;
  c00.v = acc00; c01.v = acc01; c10.v = acc10; c11.v = acc11;
  const int mrow = m0 + (wave << 4) + (half << 3);
#pragma unroll
  for (int v = 0; v < 8; ++v) {
    const int m = mrow + v;
    if (m >= M) continue;
#pragma unroll
    for (int t = 0; t < 2; ++t) {
      const int col = n0 + (t << 4) + lm;
      float val = t ? c01.f[v] : c00.f[v];
      if (MODE != 0) {
        if (col < Nvalid) val += bias[col];
        if (doRelu) val = fmaxf(val, 0.f);
      }
      O0[(size_t)m * ldO + col] = val;
      if (MODE == 0)
        O1[(size_t)m * ldO + col] = t ? c11.f[v] : c10.f[v];
    }
  }
}

// ---------------------------------------------------------------------------
// Graph kernels
// ---------------------------------------------------------------------------
__global__ void deg_kernel(const long long* __restrict__ ei,
                           float* __restrict__ deg, int E) {
  const int e = blockIdx.x * blockDim.x + threadIdx.x;
  if (e < E) unsafeAtomicAdd(&deg[(int)ei[e]], 1.0f);
}

__global__ void dinv_kernel(float* __restrict__ deg, int n) {
  const int i = blockIdx.x * blockDim.x + threadIdx.x;
  if (i < n) {
    const float d = deg[i];
    deg[i] = (d > 0.f) ? rsqrtf(d) : 0.f;
  }
}

// One 32-lane group per edge; lanes 0..24 each move a float4 (100 features).
__global__ void scatter_kernel(const float* __restrict__ P,
                               const long long* __restrict__ ei,
                               const float* __restrict__ dinv,
                               float* __restrict__ Out, int E) {
  const int gid = blockIdx.x * blockDim.x + threadIdx.x;
  const int e = gid >> 5;
  const int l = gid & 31;
  if (e >= E || l >= 25) return;
  const int s = (int)ei[e];
  const int d = (int)ei[E + e];
  const float w = -(dinv[s] * dinv[d]);
  const float4 v = *(const float4*)(P + (size_t)s * LD + (l << 2));
  float* o = Out + (size_t)d * LD + (l << 2);
  unsafeAtomicAdd(o + 0, w * v.x);
  unsafeAtomicAdd(o + 1, w * v.y);
  unsafeAtomicAdd(o + 2, w * v.z);
  unsafeAtomicAdd(o + 3, w * v.w);
}

// h1 = relu(P0 + A1 + b1) in cols [0,100), zero pad cols, written over P0.
__global__ void combine_relu_kernel(float* __restrict__ P0,
                                    const float* __restrict__ A1,
                                    const float* __restrict__ bias, int total) {
  const int i = blockIdx.x * blockDim.x + threadIdx.x;
  if (i >= total) return;
  const int c = i & (LD - 1);
  float v = 0.f;
  if (c < F_HID) v = fmaxf(P0[i] + A1[i] + bias[c], 0.f);
  P0[i] = v;
}

// ---------------------------------------------------------------------------
extern "C" void kernel_launch(void* const* d_in, const int* in_sizes, int n_in,
                              void* d_out, int out_size, void* d_ws, size_t ws_size,
                              hipStream_t stream) {
  const float*     x     = (const float*)d_in[0];
  const long long* ei    = (const long long*)d_in[1];
  const float*     w1_0  = (const float*)d_in[2];
  const float*     w1_1  = (const float*)d_in[3];
  const float*     b1    = (const float*)d_in[4];
  const float*     w2_0  = (const float*)d_in[5];
  const float*     w2_1  = (const float*)d_in[6];
  const float*     b2    = (const float*)d_in[7];
  const float*     w_lin = (const float*)d_in[8];
  const float*     b_lin = (const float*)d_in[9];
  float*           out   = (float*)d_out;

  const int N = in_sizes[0] / F_IN;
  const int E = in_sizes[1] / 2;

  char* ws = (char*)d_ws;
  const size_t nodeBuf = (size_t)N * LD * sizeof(float);
  float* dinv = (float*)ws;
  float* buf0 = (float*)(ws + (((size_t)N * 4 + 255) & ~(size_t)255));
  float* buf1 = buf0 + (size_t)N * LD;
  float* buf2 = buf1 + (size_t)N * LD;

  const dim3 blk(256);

  // Degrees -> dinv = 1/sqrt(deg) (0 if deg==0), matching the reference.
  hipMemsetAsync(dinv, 0, (size_t)N * sizeof(float), stream);
  deg_kernel<<<(E + 255) / 256, blk, 0, stream>>>(ei, dinv, E);
  dinv_kernel<<<(N + 255) / 256, blk, 0, stream>>>(dinv, N);

  // Layer 1: project first (128->100), then aggregate in 100-dim space.
  const dim3 g1((N + 127) / 128, LD / 32);
  gemm_wmma<0><<<g1, blk, 0, stream>>>(x, nullptr, w1_0, w1_1, nullptr,
                                       buf0, buf1, N, F_HID, F_IN, F_HID, LD, 0);
  hipMemsetAsync(buf2, 0, nodeBuf, stream);
  const int sblocks = (int)(((size_t)E * 32 + 255) / 256);
  scatter_kernel<<<sblocks, blk, 0, stream>>>(buf1, ei, dinv, buf2, E);
  combine_relu_kernel<<<(N * LD + 255) / 256, blk, 0, stream>>>(buf0, buf2, b1, N * LD);

  // Layer 2: aggregate in 100-dim space, then dual-input GEMM (+bias, ReLU).
  hipMemsetAsync(buf1, 0, nodeBuf, stream);
  scatter_kernel<<<sblocks, blk, 0, stream>>>(buf0, ei, dinv, buf1, E);
  gemm_wmma<1><<<g1, blk, 0, stream>>>(buf0, buf1, w2_0, w2_1, b2,
                                       buf2, nullptr, N, F_GCN, F_HID, F_GCN, LD, 1);

  // Head: out = h2 @ w_lin + b_lin.
  const dim3 g3((N + 127) / 128, F_TR / 32);
  gemm_wmma<2><<<g3, blk, 0, stream>>>(buf2, nullptr, w_lin, nullptr, b_lin,
                                       out, nullptr, N, F_TR, F_GCN, F_TR, F_TR, 0);
}
